// SelfAttention_30176440221763
// MI455X (gfx1250) — compile-verified, matrix operand order checked by reading, not run on hardware
//
#include <hip/hip_runtime.h>
#include <hip/hip_bf16.h>

// ---------------------------------------------------------------------------
// Self-attention for MI455X (gfx1250): b=2, s=2048, d_embed=1024, 16 heads x 64
// f16 WMMA (fp32 accumulate) everywhere; flash-style attention with transposed
// score tiles so softmax->operand conversion is lane-local (no LDS, no shfl).
// ---------------------------------------------------------------------------

typedef __attribute__((ext_vector_type(16))) _Float16 v16h;
typedef __attribute__((ext_vector_type(8)))  _Float16 v8h;
typedef __attribute__((ext_vector_type(8)))  float    v8f;

#define B_     2
#define S_     2048
#define E_     1024
#define H_     16
#define DH_    64
#define M_TOT  (B_ * S_)          // 4096 tokens
#define N_QKV  (3 * E_)           // 3072

__device__ __forceinline__ v8f wmma32(v16h a, v16h b, v8f c) {
  // D = A(16x32 f16) * B(32x16 f16) + C(16x16 f32)
  return __builtin_amdgcn_wmma_f32_16x16x32_f16(
      /*neg_a=*/false, a, /*neg_b=*/false, b,
      /*c_mod=*/(short)0, c, /*reuse_a=*/false, /*reuse_b=*/false);
}

// Load one 16x32 f16 operand tile (A layout; B layout assumed mirrored with
// lane = column) from a row-major [row][ld] f16 buffer.
//   lanes 0-15 : K = kBase+0..7  (v0-3) and kBase+16..23 (v4-7)
//   lanes 16-31: K = kBase+8..15 (v0-3) and kBase+24..31 (v4-7)
__device__ __forceinline__ v16h load_tile(const _Float16* base, int row, int ld,
                                          int kBase, int lane) {
  const int half8 = (lane & 16) ? 8 : 0;
  const _Float16* p = base + (size_t)row * ld + kBase + half8;
  v8h lo = *(const v8h*)p;
  v8h hi = *(const v8h*)(p + 16);
  return __builtin_shufflevector(lo, hi, 0, 1, 2, 3, 4, 5, 6, 7,
                                 8, 9, 10, 11, 12, 13, 14, 15);
}

// --------------------------- convert kernels -------------------------------

__global__ void cvt_f32_to_f16(const float* __restrict__ in,
                               _Float16* __restrict__ out, int n) {
  int i = blockIdx.x * blockDim.x + threadIdx.x;
  if (i < n) out[i] = (_Float16)in[i];
}

// out[n*K + k] = (f16) in[k*N + n]   (transpose so B tiles load contiguously)
__global__ void cvt_transpose_f16(const float* __restrict__ in,
                                  _Float16* __restrict__ out, int K, int N) {
  int i = blockIdx.x * blockDim.x + threadIdx.x;
  if (i < K * N) {
    int k = i / N, n = i % N;
    out[(size_t)n * K + k] = (_Float16)in[i];
  }
}

// ------------------------------ QKV GEMM -----------------------------------
// C[4096,3072] = Xh[4096,1024] * WinT[3072,1024]^T + b_in
// One wave (32 threads) computes a 64x64 tile; Q pre-scaled by 1/sqrt(64).
__global__ void __launch_bounds__(32)
qkv_gemm(const _Float16* __restrict__ Xh, const _Float16* __restrict__ WinT,
         const float* __restrict__ b_in, _Float16* __restrict__ Qb,
         _Float16* __restrict__ Kb, _Float16* __restrict__ Vt) {
  const int lane = threadIdx.x;
  const int col  = lane & 15;
  const int half = lane >> 4;
  const int nBase = blockIdx.x * 64;
  const int mBase = blockIdx.y * 64;

  v8f acc[4][4];
  for (int i = 0; i < 4; ++i)
    for (int j = 0; j < 4; ++j) acc[i][j] = {};

  for (int kk = 0; kk < E_; kk += 32) {
    v16h a[4], b[4];
#pragma unroll
    for (int mi = 0; mi < 4; ++mi)
      a[mi] = load_tile(Xh, mBase + mi * 16 + col, E_, kk, lane);
#pragma unroll
    for (int ni = 0; ni < 4; ++ni)
      b[ni] = load_tile(WinT, nBase + ni * 16 + col, E_, kk, lane);
    // hint next K-slab into cache (global_prefetch_b8)
    __builtin_prefetch(Xh + (size_t)(mBase + col) * E_ + kk + 32, 0, 3);
    __builtin_prefetch(WinT + (size_t)(nBase + col) * E_ + kk + 32, 0, 3);
#pragma unroll
    for (int mi = 0; mi < 4; ++mi)
#pragma unroll
      for (int ni = 0; ni < 4; ++ni)
        acc[mi][ni] = wmma32(a[mi], b[ni], acc[mi][ni]);
  }

  // Epilogue. C layout: vgpr j holds M = j (lanes<16) / j+8 (lanes>=16), N=col.
  for (int ni = 0; ni < 4; ++ni) {
    const int n = nBase + ni * 16 + col;
    const float bias = b_in[n];
    for (int mi = 0; mi < 4; ++mi) {
      const int tok0 = mBase + mi * 16 + half * 8;
      v8f c = acc[mi][ni];
      if (n < E_) {                       // Q: row-major [b,h,s,d], pre-scaled
        const int h = n >> 6, d = n & 63;
        for (int j = 0; j < 8; ++j) {
          const int tok = tok0 + j, bb = tok >> 11, ss = tok & (S_ - 1);
          Qb[(((size_t)(bb * H_ + h)) * S_ + ss) * DH_ + d] =
              (_Float16)((c[j] + bias) * 0.125f);
        }
      } else if (n < 2 * E_) {            // K: row-major [b,h,s,d]
        const int n2 = n - E_, h = n2 >> 6, d = n2 & 63;
        for (int j = 0; j < 8; ++j) {
          const int tok = tok0 + j, bb = tok >> 11, ss = tok & (S_ - 1);
          Kb[(((size_t)(bb * H_ + h)) * S_ + ss) * DH_ + d] =
              (_Float16)(c[j] + bias);
        }
      } else {                            // V: transposed [b,h,d,s], packed b128
        const int n2 = n - 2 * E_, h = n2 >> 6, d = n2 & 63;
        const int bb = tok0 >> 11, ss0 = tok0 & (S_ - 1);
        v8h pk;
        for (int j = 0; j < 8; ++j) pk[j] = (_Float16)(c[j] + bias);
        *(v8h*)(Vt + (((size_t)(bb * H_ + h)) * DH_ + d) * S_ + ss0) = pk;
      }
    }
  }
}

// ----------------------------- Attention -----------------------------------
// One wave = 16 queries of one (b,h), streaming all 2048 keys in blocks of 32.
// S^T = K * Q^T  (M=k rows, N=q cols) -> P^T packs lane-locally into the
// B operand of O^T = V^T * P^T. Row stats (m,l,alpha) are per-lane (q = lane%16).
__global__ void __launch_bounds__(32)
attn_kernel(const _Float16* __restrict__ Qb, const _Float16* __restrict__ Kb,
            const _Float16* __restrict__ Vt, _Float16* __restrict__ At) {
  const int lane = threadIdx.x;
  const int col  = lane & 15;
  const int half = lane >> 4;
  const int bh = blockIdx.x >> 7;          // 0..31  (b*16 + h)
  const int qBase = (blockIdx.x & 127) * 16;

  const _Float16* Qh = Qb + (size_t)bh * S_ * DH_;
  const _Float16* Kh = Kb + (size_t)bh * S_ * DH_;
  const _Float16* Vh = Vt + (size_t)bh * DH_ * S_;

  // Q^T as B operand (32d x 16q), d = 0..31 and 32..63; loaded once.
  const v16h bq0 = load_tile(Qh, qBase + col, DH_, 0, lane);
  const v16h bq1 = load_tile(Qh, qBase + col, DH_, 32, lane);

  v8f o0 = {}, o1 = {}, o2 = {}, o3 = {};   // O^T: 4 tiles of (16d x 16q)
  float m = -1e30f, l = 0.0f;

  for (int sk = 0; sk < S_; sk += 32) {
    // --- transposed score tiles (k rows sk..+15 and sk+16..+31) ---
    v8f st0 = {}, st1 = {};
    v16h ak;
    ak = load_tile(Kh, sk + col, DH_, 0, lane);       st0 = wmma32(ak, bq0, st0);
    ak = load_tile(Kh, sk + col, DH_, 32, lane);      st0 = wmma32(ak, bq1, st0);
    ak = load_tile(Kh, sk + 16 + col, DH_, 0, lane);  st1 = wmma32(ak, bq0, st1);
    ak = load_tile(Kh, sk + 16 + col, DH_, 32, lane); st1 = wmma32(ak, bq1, st1);

    // --- online softmax: per-lane holds 16 of the 32 keys of row q=col ---
    float mb = st0[0];
#pragma unroll
    for (int j = 1; j < 8; ++j) mb = fmaxf(mb, st0[j]);
#pragma unroll
    for (int j = 0; j < 8; ++j) mb = fmaxf(mb, st1[j]);
    mb = fmaxf(mb, __shfl_xor(mb, 16, 32));
    const float mn = fmaxf(m, mb);
    const float alpha = __expf(m - mn);
    m = mn;

    v8f p0, p1;
    float ls = 0.0f;
#pragma unroll
    for (int j = 0; j < 8; ++j) { p0[j] = __expf(st0[j] - mn); ls += p0[j]; }
#pragma unroll
    for (int j = 0; j < 8; ++j) { p1[j] = __expf(st1[j] - mn); ls += p1[j]; }
    ls += __shfl_xor(ls, 16, 32);
    l = l * alpha + ls;

    // --- lane-local pack of P^T into WMMA B operand (32k x 16q) ---
    v16h bp;
#pragma unroll
    for (int j = 0; j < 8; ++j) {
      bp[j]     = (_Float16)p0[j];
      bp[8 + j] = (_Float16)p1[j];
    }

    o0 = o0 * alpha; o1 = o1 * alpha; o2 = o2 * alpha; o3 = o3 * alpha;

    // --- O^T += V^T(16d x 32k) * P^T ---
    v16h av;
    av = load_tile(Vh, 0  + col, S_, sk, lane); o0 = wmma32(av, bp, o0);
    av = load_tile(Vh, 16 + col, S_, sk, lane); o1 = wmma32(av, bp, o1);
    av = load_tile(Vh, 32 + col, S_, sk, lane); o2 = wmma32(av, bp, o2);
    av = load_tile(Vh, 48 + col, S_, sk, lane); o3 = wmma32(av, bp, o3);
  }

  // Normalize and store: lane holds q=col, d = t*16 + half*8 + j (contiguous).
  const float rl = 1.0f / l;
  const int b = bh >> 4, h = bh & 15;
  const int q = qBase + col;
  _Float16* outp = At + ((size_t)(b * S_ + q)) * E_ + h * DH_ + half * 8;
  v8f ot[4] = {o0, o1, o2, o3};
#pragma unroll
  for (int t = 0; t < 4; ++t) {
    v8h pk;
#pragma unroll
    for (int j = 0; j < 8; ++j) pk[j] = (_Float16)(ot[t][j] * rl);
    *(v8h*)(outp + t * 16) = pk;
  }
}

// --------------------------- Output projection -----------------------------
// out[4096,1024] = At[4096,1024] * WoutT[1024,1024]^T + b_out  (fp32 out)
__global__ void __launch_bounds__(32)
out_gemm(const _Float16* __restrict__ At, const _Float16* __restrict__ WoutT,
         const float* __restrict__ b_out, float* __restrict__ out) {
  const int lane = threadIdx.x;
  const int col  = lane & 15;
  const int half = lane >> 4;
  const int nBase = blockIdx.x * 64;
  const int mBase = blockIdx.y * 64;

  v8f acc[4][4];
  for (int i = 0; i < 4; ++i)
    for (int j = 0; j < 4; ++j) acc[i][j] = {};

  for (int kk = 0; kk < E_; kk += 32) {
    v16h a[4], b[4];
#pragma unroll
    for (int mi = 0; mi < 4; ++mi)
      a[mi] = load_tile(At, mBase + mi * 16 + col, E_, kk, lane);
#pragma unroll
    for (int ni = 0; ni < 4; ++ni)
      b[ni] = load_tile(WoutT, nBase + ni * 16 + col, E_, kk, lane);
    __builtin_prefetch(At + (size_t)(mBase + col) * E_ + kk + 32, 0, 3);
#pragma unroll
    for (int mi = 0; mi < 4; ++mi)
#pragma unroll
      for (int ni = 0; ni < 4; ++ni)
        acc[mi][ni] = wmma32(a[mi], b[ni], acc[mi][ni]);
  }

  for (int ni = 0; ni < 4; ++ni) {
    const int n = nBase + ni * 16 + col;
    const float bias = b_out[n];
    for (int mi = 0; mi < 4; ++mi) {
      const int tok0 = mBase + mi * 16 + half * 8;
      v8f c = acc[mi][ni];
      for (int j = 0; j < 8; ++j)
        out[(size_t)(tok0 + j) * E_ + n] = c[j] + bias;
    }
  }
}

// ------------------------------- launcher ----------------------------------

extern "C" void kernel_launch(void* const* d_in, const int* in_sizes, int n_in,
                              void* d_out, int out_size, void* d_ws,
                              size_t ws_size, hipStream_t stream) {
  const float* x     = (const float*)d_in[0];   // (2,2048,1024)
  const float* w_in  = (const float*)d_in[1];   // (1024,3072)
  const float* b_in  = (const float*)d_in[2];   // (3072,)
  const float* w_out = (const float*)d_in[3];   // (1024,1024)
  const float* b_out = (const float*)d_in[4];   // (1024,)
  float* out = (float*)d_out;

  _Float16* ws = (_Float16*)d_ws;
  _Float16* Xh    = ws;                              // 4096*1024
  _Float16* WinT  = Xh    + (size_t)M_TOT * E_;      // 3072*1024 (transposed)
  _Float16* WoutT = WinT  + (size_t)N_QKV * E_;      // 1024*1024 (transposed)
  _Float16* Qb    = WoutT + (size_t)E_ * E_;         // [b,h,s,d]
  _Float16* Kb    = Qb    + (size_t)B_ * H_ * S_ * DH_;
  _Float16* Vt    = Kb    + (size_t)B_ * H_ * S_ * DH_;  // [b,h,d,s]
  _Float16* At    = Vt    + (size_t)B_ * H_ * S_ * DH_;  // [b,s,h*d]
  (void)ws_size; (void)in_sizes; (void)n_in; (void)out_size;

  cvt_f32_to_f16<<<(M_TOT * E_ + 255) / 256, 256, 0, stream>>>(x, Xh,
                                                               M_TOT * E_);
  cvt_transpose_f16<<<(E_ * N_QKV + 255) / 256, 256, 0, stream>>>(w_in, WinT,
                                                                  E_, N_QKV);
  cvt_transpose_f16<<<(E_ * E_ + 255) / 256, 256, 0, stream>>>(w_out, WoutT,
                                                               E_, E_);

  qkv_gemm<<<dim3(N_QKV / 64, M_TOT / 64), 32, 0, stream>>>(Xh, WinT, b_in,
                                                            Qb, Kb, Vt);

  attn_kernel<<<B_ * H_ * (S_ / 16), 32, 0, stream>>>(Qb, Kb, Vt, At);

  out_gemm<<<dim3(E_ / 64, M_TOT / 64), 32, 0, stream>>>(At, WoutT, b_out, out);
}
